// MultiHeadAttention_unboxed_51402168598918
// MI455X (gfx1250) — compile-verified, matrix operand order checked by reading, not run on hardware
//
#include <hip/hip_runtime.h>

// ---------------------------------------------------------------------------
// MHA block for MI455X (gfx1250, wave32).  All GEMMs use
// v_wmma_f32_16x16x32_bf16 (fp32 accumulate).  Attention is flash-style
// (online softmax), P-tiles staged through LDS for layout conversion.
// Inner GEMM loops are split into load-phase / compute-phase so the
// scheduler can overlap global_load_b128 latency with WMMA issue
// (partial s_wait_loadcnt instead of waitcnt-0 before every WMMA).
// ---------------------------------------------------------------------------

#define BB 2
#define SS 2048
#define DD 1024
#define HH 16
#define DKV 64

typedef __attribute__((ext_vector_type(16))) __bf16 v16bf;
typedef __attribute__((ext_vector_type(8)))  float  v8f;

union FragU { v16bf v; uint4 q[2]; };

__device__ __forceinline__ unsigned short f2bf(float f) {
  unsigned int u = __float_as_uint(f);
  u += 0x7FFFu + ((u >> 16) & 1u);          // round-to-nearest-even
  return (unsigned short)(u >> 16);
}

// Load one WMMA bf16 fragment (A or B) for this lane from a row-major
// [16 x 32] tile whose row for this lane starts at `rowbase`.
// Lane layout per CDNA5 ISA 7.12.2:
//   lanes 0-15 : K in {0..7} and {16..23}
//   lanes 16-31: K in {8..15} and {24..31}
__device__ __forceinline__ v16bf load_frag(const unsigned short* rowbase, int lane) {
  const unsigned short* p = rowbase + ((lane & 16) ? 8 : 0);
  FragU f;
  f.q[0] = *reinterpret_cast<const uint4*>(p);
  f.q[1] = *reinterpret_cast<const uint4*>(p + 16);
  return f.v;
}

#define WMMA_BF16(a, b, c) \
  __builtin_amdgcn_wmma_f32_16x16x32_bf16(false, (a), false, (b), (short)0, (c), false, false)

// ---------------------------------------------------------------------------
// fp32 -> bf16 elementwise convert
// ---------------------------------------------------------------------------
__global__ void cvt_kernel(const float* __restrict__ src,
                           unsigned short* __restrict__ dst, int n) {
  int i = blockIdx.x * blockDim.x + threadIdx.x;
  if (i < n) dst[i] = f2bf(src[i]);
}

// fp32 [G,R,C] -> bf16 [G,C,R] (transpose inner two dims)
__global__ void cvtT_kernel(const float* __restrict__ src,
                            unsigned short* __restrict__ dst,
                            int G, int R, int C) {
  int i = blockIdx.x * blockDim.x + threadIdx.x;
  int n = G * R * C;
  if (i >= n) return;
  int g = i / (R * C);
  int rem = i - g * (R * C);
  int r = rem / C, c = rem - r * C;
  dst[(size_t)g * R * C + (size_t)c * R + r] = f2bf(src[i]);
}

// ---------------------------------------------------------------------------
// Per-head projection: OUT[bh] = X[b] @ W[h]  ([S,D] x [D,64])
// X : bf16 [B,S,D] row-major, WT : bf16 [H,64,D] (pre-transposed)
// transposeOut=0 -> OUT [B*H, S, 64]; =1 -> OUT [B*H, 64, S]
// grid (B*H, S/64), block 128 (4 waves, 16 rows each)
// ---------------------------------------------------------------------------
__global__ void proj_kernel(const unsigned short* __restrict__ X,
                            const unsigned short* __restrict__ WT,
                            unsigned short* __restrict__ OUT,
                            float scale, int transposeOut) {
  const int lane = threadIdx.x & 31;
  const int wave = threadIdx.x >> 5;
  const int bh = blockIdx.x;
  const int b = bh / HH, h = bh % HH;
  const int s_base = blockIdx.y * 64 + wave * 16;

  const unsigned short* Xrow =
      X + ((size_t)b * SS + (size_t)(s_base + (lane & 15))) * DD;
  const unsigned short* Wrow = WT + ((size_t)h * DKV + (lane & 15)) * DD;

  v8f acc[4] = {};
#pragma unroll 2
  for (int k0 = 0; k0 < DD; k0 += 32) {
    __builtin_prefetch(Xrow + k0 + 64, 0, 1);
    // ---- load phase ----
    v16bf a = load_frag(Xrow + k0, lane);
    v16bf bf[4];
#pragma unroll
    for (int nt = 0; nt < 4; ++nt)
      bf[nt] = load_frag(Wrow + (size_t)nt * 16 * DD + k0, lane);
    // ---- compute phase ----
#pragma unroll
    for (int nt = 0; nt < 4; ++nt)
      acc[nt] = WMMA_BF16(a, bf[nt], acc[nt]);
  }
  const int rbase = (lane & 16) ? 8 : 0;
  const int col0 = lane & 15;
#pragma unroll
  for (int nt = 0; nt < 4; ++nt) {
#pragma unroll
    for (int r = 0; r < 8; ++r) {
      int srow = s_base + rbase + r;
      int col = nt * 16 + col0;
      unsigned short val = f2bf(acc[nt][r] * scale);
      if (transposeOut)
        OUT[(size_t)bh * DKV * SS + (size_t)col * SS + srow] = val;
      else
        OUT[(size_t)bh * SS * DKV + (size_t)srow * DKV + col] = val;
    }
  }
}

// ---------------------------------------------------------------------------
// Flash attention per (b,h): scores = QH @ KH^T (QH pre-scaled by 1/sqrt(dk)),
// masked online softmax, O = P @ V.  grid (B*H, S/64), block 128 (4 waves).
// QH/KH : bf16 [B*H, S, 64]; VT : bf16 [B*H, 64, S]; AOUT : bf16 [B, S, H*64]
// V fragments are prefetched before the softmax so their latency hides
// behind the exp/shuffle VALU work.
// ---------------------------------------------------------------------------
__global__ void attn_kernel(const unsigned short* __restrict__ QH,
                            const unsigned short* __restrict__ KH,
                            const unsigned short* __restrict__ VT,
                            const int* __restrict__ MASK,
                            unsigned short* __restrict__ AOUT) {
  __shared__ __align__(16) unsigned short Pld[4][16][32];

  const int lane = threadIdx.x & 31;
  const int wave = threadIdx.x >> 5;
  const int bh = blockIdx.x;
  const int b = bh >> 4, h = bh & 15;
  const int s_base = blockIdx.y * 64 + wave * 16;
  const int rbase = (lane & 16) ? 8 : 0;
  const int col = lane & 15;

  // Q fragments for the whole dk=64 (two 32-wide K chunks), loaded once.
  const unsigned short* Qrow =
      QH + ((size_t)bh * SS + (size_t)(s_base + col)) * DKV;
  v16bf qf0 = load_frag(Qrow, lane);
  v16bf qf1 = load_frag(Qrow + 32, lane);

  v8f o[4] = {};
  float m[8], l[8];
#pragma unroll
  for (int r = 0; r < 8; ++r) { m[r] = -3.0e38f; l[r] = 0.0f; }

  const int* maskbase = MASK + (size_t)b * SS * SS;
  const unsigned short* Krow = KH + ((size_t)bh * SS + col) * DKV;
  const unsigned short* Vrow = VT + ((size_t)bh * DKV + col) * SS;

  for (int t0 = 0; t0 < SS; t0 += 32) {
    // ---- load phase: K fragments for keys t0..t0+31, V frags for same ----
    v16bf kb0 = load_frag(Krow + (size_t)t0 * DKV, lane);
    v16bf kb1 = load_frag(Krow + (size_t)t0 * DKV + 32, lane);
    v16bf kb2 = load_frag(Krow + (size_t)(t0 + 16) * DKV, lane);
    v16bf kb3 = load_frag(Krow + (size_t)(t0 + 16) * DKV + 32, lane);
    v16bf vb[4];
#pragma unroll
    for (int nt = 0; nt < 4; ++nt)
      vb[nt] = load_frag(Vrow + (size_t)nt * 16 * SS + t0, lane);

    // ---- scores: two 16x16 tiles ----
    v8f s0 = {}, s1 = {};
    s0 = WMMA_BF16(qf0, kb0, s0);
    s0 = WMMA_BF16(qf1, kb1, s0);
    s1 = WMMA_BF16(qf0, kb2, s1);
    s1 = WMMA_BF16(qf1, kb3, s1);

    // ---- mask + online softmax (C layout: reg r -> row rbase+r, N = col) ----
#pragma unroll
    for (int r = 0; r < 8; ++r) {
      const int* mrow = maskbase + (size_t)(s_base + rbase + r) * SS + t0;
      float v0 = (mrow[col]      != 0) ? s0[r] : -1.0e9f;
      float v1 = (mrow[col + 16] != 0) ? s1[r] : -1.0e9f;
      float mx = fmaxf(v0, v1);
      mx = fmaxf(mx, __shfl_xor(mx, 1));
      mx = fmaxf(mx, __shfl_xor(mx, 2));
      mx = fmaxf(mx, __shfl_xor(mx, 4));
      mx = fmaxf(mx, __shfl_xor(mx, 8));
      float mnew  = fmaxf(m[r], mx);
      float alpha = __expf(m[r] - mnew);
      float p0 = __expf(v0 - mnew);
      float p1 = __expf(v1 - mnew);
      float rs = p0 + p1;
      rs += __shfl_xor(rs, 1);
      rs += __shfl_xor(rs, 2);
      rs += __shfl_xor(rs, 4);
      rs += __shfl_xor(rs, 8);
      l[r] = l[r] * alpha + rs;
      m[r] = mnew;
#pragma unroll
      for (int nt = 0; nt < 4; ++nt) o[nt][r] = o[nt][r] * alpha;
      // P tile -> LDS (wave-private region; DS is in-order per wave)
      Pld[wave][rbase + r][col]      = f2bf(p0);
      Pld[wave][rbase + r][col + 16] = f2bf(p1);
    }
    asm volatile("s_wait_dscnt 0" ::: "memory");
    // ---- O += P @ V ----
    v16bf pf = load_frag(&Pld[wave][col][0], lane);
#pragma unroll
    for (int nt = 0; nt < 4; ++nt)
      o[nt] = WMMA_BF16(pf, vb[nt], o[nt]);
  }

  // ---- normalize and write concat-head output ----
#pragma unroll
  for (int r = 0; r < 8; ++r) {
    float inv = 1.0f / l[r];
    int srow = s_base + rbase + r;
#pragma unroll
    for (int nt = 0; nt < 4; ++nt) {
      int dv = nt * 16 + col;
      AOUT[((size_t)b * SS + srow) * (HH * DKV) + h * DKV + dv] =
          f2bf(o[nt][r] * inv);
    }
  }
}

// ---------------------------------------------------------------------------
// FC + residual: OUT[r,n] = sum_k A[r,k]*Wfc[k,n] + RES[r,n]   (fp32 out)
// A : bf16 [4096,1024]; WT : bf16 [1024(n),1024(k)] pre-transposed.
// grid 256 blocks (16 rows each), block 256 (8 waves x 128 cols)
// ---------------------------------------------------------------------------
__global__ void fc_kernel(const unsigned short* __restrict__ A,
                          const unsigned short* __restrict__ WT,
                          const float* __restrict__ RES,
                          float* __restrict__ OUT) {
  const int lane = threadIdx.x & 31;
  const int wave = threadIdx.x >> 5;
  const int row0 = blockIdx.x * 16;
  const int n0 = wave * 128;

  const unsigned short* Arow = A + (size_t)(row0 + (lane & 15)) * DD;
  const unsigned short* Wrow = WT + (size_t)(n0 + (lane & 15)) * DD;

  v8f acc[8] = {};
  for (int k0 = 0; k0 < DD; k0 += 32) {
    __builtin_prefetch(Arow + k0 + 64, 0, 1);
    // ---- load phase ----
    v16bf a = load_frag(Arow + k0, lane);
    v16bf bf[8];
#pragma unroll
    for (int nt = 0; nt < 8; ++nt)
      bf[nt] = load_frag(Wrow + (size_t)nt * 16 * DD + k0, lane);
    // ---- compute phase ----
#pragma unroll
    for (int nt = 0; nt < 8; ++nt)
      acc[nt] = WMMA_BF16(a, bf[nt], acc[nt]);
  }
  const int rbase = (lane & 16) ? 8 : 0;
  const int col0 = lane & 15;
#pragma unroll
  for (int nt = 0; nt < 8; ++nt) {
#pragma unroll
    for (int r = 0; r < 8; ++r) {
      size_t idx = (size_t)(row0 + rbase + r) * DD + n0 + nt * 16 + col0;
      OUT[idx] = acc[nt][r] + RES[idx];
    }
  }
}

// ---------------------------------------------------------------------------
// LayerNorm over D=1024 per row; one block (256 thr) per row.
// ---------------------------------------------------------------------------
__global__ void ln_kernel(const float* __restrict__ X,
                          const float* __restrict__ W,
                          const float* __restrict__ Bv,
                          float* __restrict__ OUT) {
  __shared__ float red[8];
  const int row = blockIdx.x;
  const int tid = threadIdx.x;
  const float* xr = X + (size_t)row * DD;

  float v[4];
  float s = 0.0f;
#pragma unroll
  for (int i = 0; i < 4; ++i) { v[i] = xr[tid + i * 256]; s += v[i]; }
  for (int off = 1; off < 32; off <<= 1) s += __shfl_xor(s, off);
  if ((tid & 31) == 0) red[tid >> 5] = s;
  __syncthreads();
  float mu = (red[0] + red[1] + red[2] + red[3] +
              red[4] + red[5] + red[6] + red[7]) * (1.0f / DD);
  float vs = 0.0f;
#pragma unroll
  for (int i = 0; i < 4; ++i) { float d = v[i] - mu; vs += d * d; }
  for (int off = 1; off < 32; off <<= 1) vs += __shfl_xor(vs, off);
  __syncthreads();
  if ((tid & 31) == 0) red[tid >> 5] = vs;
  __syncthreads();
  float var = (red[0] + red[1] + red[2] + red[3] +
               red[4] + red[5] + red[6] + red[7]) * (1.0f / DD);
  float rstd = rsqrtf(var + 1e-6f);
#pragma unroll
  for (int i = 0; i < 4; ++i) {
    int c = tid + i * 256;
    OUT[(size_t)row * DD + c] = (v[i] - mu) * rstd * W[c] + Bv[c];
  }
}

// ---------------------------------------------------------------------------
extern "C" void kernel_launch(void* const* d_in, const int* in_sizes, int n_in,
                              void* d_out, int out_size, void* d_ws, size_t ws_size,
                              hipStream_t stream) {
  const float* q    = (const float*)d_in[0];
  const float* k    = (const float*)d_in[1];
  const float* v    = (const float*)d_in[2];
  const int*   mask = (const int*)d_in[3];
  const float* Wq   = (const float*)d_in[4];
  const float* Wk   = (const float*)d_in[5];
  const float* Wv   = (const float*)d_in[6];
  const float* Wfc  = (const float*)d_in[7];
  const float* lnw  = (const float*)d_in[8];
  const float* lnb  = (const float*)d_in[9];
  float* out = (float*)d_out;

  char* ws = (char*)d_ws;
  const size_t SZ_ACT  = (size_t)BB * SS * DD * 2;      //  8 MB bf16 act
  const size_t SZ_W    = (size_t)HH * DD * DKV * 2;     //  2 MB bf16 head W
  const size_t SZ_HEAD = (size_t)BB * HH * SS * DKV * 2;//  8 MB bf16 per-head
  size_t off = 0;
  unsigned short* qbf  = (unsigned short*)(ws + off); off += SZ_ACT;
  unsigned short* kbf  = (unsigned short*)(ws + off); off += SZ_ACT;
  unsigned short* vbf  = (unsigned short*)(ws + off); off += SZ_ACT;
  unsigned short* WqT  = (unsigned short*)(ws + off); off += SZ_W;
  unsigned short* WkT  = (unsigned short*)(ws + off); off += SZ_W;
  unsigned short* WvT  = (unsigned short*)(ws + off); off += SZ_W;
  unsigned short* WfcT = (unsigned short*)(ws + off); off += (size_t)DD * DD * 2;
  unsigned short* qh   = (unsigned short*)(ws + off); off += SZ_HEAD;
  unsigned short* kh   = (unsigned short*)(ws + off); off += SZ_HEAD;
  unsigned short* vhT  = (unsigned short*)(ws + off); off += SZ_HEAD;
  unsigned short* aout = (unsigned short*)(ws + off); off += SZ_ACT;
  float*          fco  = (float*)(ws + off);          off += (size_t)BB * SS * DD * 4;

  const int nAct = BB * SS * DD;       // 4194304
  const int nW   = HH * DD * DKV;      // 1048576
  const int nFc  = DD * DD;            // 1048576

  cvt_kernel<<<(nAct + 255) / 256, 256, 0, stream>>>(q, qbf, nAct);
  cvt_kernel<<<(nAct + 255) / 256, 256, 0, stream>>>(k, kbf, nAct);
  cvt_kernel<<<(nAct + 255) / 256, 256, 0, stream>>>(v, vbf, nAct);
  cvtT_kernel<<<(nW + 255) / 256, 256, 0, stream>>>(Wq, WqT, HH, DD, DKV);
  cvtT_kernel<<<(nW + 255) / 256, 256, 0, stream>>>(Wk, WkT, HH, DD, DKV);
  cvtT_kernel<<<(nW + 255) / 256, 256, 0, stream>>>(Wv, WvT, HH, DD, DKV);
  cvtT_kernel<<<(nFc + 255) / 256, 256, 0, stream>>>(Wfc, WfcT, 1, DD, DD);

  dim3 gProj(BB * HH, SS / 64);
  proj_kernel<<<gProj, 128, 0, stream>>>(qbf, WqT, qh, 0.125f, 0); // 1/sqrt(64)
  proj_kernel<<<gProj, 128, 0, stream>>>(kbf, WkT, kh, 1.0f, 0);
  proj_kernel<<<gProj, 128, 0, stream>>>(vbf, WvT, vhT, 1.0f, 1);

  attn_kernel<<<dim3(BB * HH, SS / 64), 128, 0, stream>>>(qh, kh, vhT, mask, aout);

  fc_kernel<<<(BB * SS) / 16, 256, 0, stream>>>(aout, WfcT, q, fco);
  ln_kernel<<<BB * SS, 256, 0, stream>>>(fco, lnw, lnb, out);
}
